// TensHash_22806276342196
// MI455X (gfx1250) — 8-bit matrix operand order repaired and checked by reading, compiled, not run on hardware
//
#include <hip/hip_runtime.h>
#include <hip/hip_bf16.h>
#include <stdint.h>

#define ROUNDS 64
#define HIDDEN 256
#define BATCH  65536
#define WAVES_PER_BLOCK 4
#define ROWS_PER_WAVE 32
#define THREADS (WAVES_PER_BLOCK * 32)
#define MATBYTES (HIDDEN * HIDDEN)

#define SCALE  0x1p-9f
#define MAGIC  24576.0f

typedef __attribute__((ext_vector_type(16))) int   v16i;
typedef __attribute__((ext_vector_type(8)))  float v8f;

__device__ __forceinline__ unsigned fp8_enc_w(float w) {
    return (w == 0.0f) ? 0u : (w > 0.0f ? 0x38u : 0xB8u);
}

__global__ __launch_bounds__(256) void pack_matB(const float* __restrict__ M,
                                                 unsigned int* __restrict__ matB) {
    int idx  = blockIdx.x * 256 + threadIdx.x;
    int v    = idx & 15;
    int lane = (idx >> 4) & 31;
    int h    = (idx >> 9) & 1;
    int j    = (idx >> 10) & 15;
    int r    = idx >> 14;
    int n    = j * 16 + (lane & 15);
    int lh   = lane >> 4;
    int g    = v >> 2;
    unsigned d = 0;
#pragma unroll
    for (int b = 0; b < 4; ++b) {
        int kk = ((v & 3) << 2) | b;
        int k  = h * 128 + (((v >> 1) << 4) | (lh << 3) | ((v & 1) << 2) | b);
        float w = M[((size_t)r * HIDDEN + n) * HIDDEN + k];
        d |= fp8_enc_w(w) << (8 * b);
    }
    matB[idx] = d;
}

__global__ __launch_bounds__(THREADS) void tenshash_rounds(const float* __restrict__ state,
                                                           const float* __restrict__ noise,
                                                           const unsigned char* __restrict__ matB,
                                                           float* __restrict__ out) {
    extern __shared__ unsigned char lds[];
    const int lane = threadIdx.x & 31;
    const int wave = threadIdx.x >> 5;
    unsigned char* ldsS = lds + 2 * MATBYTES + (wave << 13);

    typedef __attribute__((address_space(3))) unsigned char lds_as3_t;
    const unsigned ldsBase = (unsigned)(unsigned long long)(lds_as3_t*)lds;

    const int tileBase = (blockIdx.x * WAVES_PER_BLOCK + wave) * ROWS_PER_WAVE;
    const int m  = lane & 15;
    const int lh = lane >> 4;

    const unsigned voff = threadIdx.x * 16u;

    {
        unsigned dst = ldsBase + voff;
#pragma unroll
        for (int i = 0; i < 32; ++i)
            asm volatile("global_load_async_to_lds_b128 %0, %1, %2 offset:%3"
                         :: "v"(dst), "v"(voff), "s"(matB), "n"(i * (THREADS * 16)) : "memory");
    }

    v16i A[2][2];
#pragma unroll
    for (int t = 0; t < 2; ++t) {
        const float* srow = state + (size_t)(tileBase + t * 16 + m) * HIDDEN;
#pragma unroll
        for (int H = 0; H < 2; ++H) {
#pragma unroll
            for (int e = 0; e < 16; ++e) {
                int v8 = e & 7;
                int k  = H * 128 + ((e >> 3) << 6) +
                         (((v8 >> 1) << 4) | (lh << 3) | ((v8 & 1) << 2));
                unsigned b0 = __builtin_nontemporal_load(&srow[k + 0]) != 0.0f ? 1u : 0u;
                unsigned b1 = __builtin_nontemporal_load(&srow[k + 1]) != 0.0f ? 1u : 0u;
                unsigned b2 = __builtin_nontemporal_load(&srow[k + 2]) != 0.0f ? 1u : 0u;
                unsigned b3 = __builtin_nontemporal_load(&srow[k + 3]) != 0.0f ? 1u : 0u;
                A[t][H][e] = (int)(b0 | (b1 << 8) | (b2 << 16) | (b3 << 24));
            }
        }
    }

    asm volatile("s_wait_asynccnt 0x0" ::: "memory");
    __syncthreads();

#pragma unroll 1
    for (int r = 0; r < ROUNDS; ++r) {
        unsigned char* ldsM = lds + (unsigned)(r & 1) * MATBYTES;

        if (r + 1 < ROUNDS) {
            const unsigned char* src = matB + (size_t)(r + 1) * MATBYTES;
            unsigned dst = ldsBase + (unsigned)((r + 1) & 1) * MATBYTES + voff;
#pragma unroll
            for (int i = 0; i < 32; ++i)
                asm volatile("global_load_async_to_lds_b128 %0, %1, %2 offset:%3"
                             :: "v"(dst), "v"(voff), "s"(src), "n"(i * (THREADS * 16)) : "memory");
        }

        const float* nrow = noise + (size_t)r * HIDDEN;
#pragma unroll
        for (int j = 0; j < 16; ++j) {
            v16i B0 = *(const v16i*)(ldsM + (size_t)(j * 2 + 0) * 2048 + (size_t)lane * 64);
            v16i B1 = *(const v16i*)(ldsM + (size_t)(j * 2 + 1) * 2048 + (size_t)lane * 64);

            float nzf = nrow[j * 16 + m] * SCALE;
            v8f cinit = {nzf, nzf, nzf, nzf, nzf, nzf, nzf, nzf};

#pragma unroll
            for (int t = 0; t < 2; ++t) {
                v8f acc = __builtin_amdgcn_wmma_f32_16x16x128_fp8_fp8(A[t][0], B0, (short)0, cinit, false, false);
                acc     = __builtin_amdgcn_wmma_f32_16x16x128_fp8_fp8(A[t][1], B1, (short)0, acc,   false, false);

                unsigned char* dcol = ldsS + (size_t)(t * 16 + (lh << 3)) * 256 + j * 16 + m;
#pragma unroll
                for (int v = 0; v < 8; ++v) {
                    unsigned bit = __float_as_uint(acc[v] + MAGIC) & 1u;
                    dcol[v * 256] = (unsigned char)bit;
                }
            }
        }

        asm volatile("s_wait_dscnt 0x0" ::: "memory");

#pragma unroll
        for (int t = 0; t < 2; ++t) {
#pragma unroll
            for (int H = 0; H < 2; ++H) {
#pragma unroll
                for (int p = 0; p < 8; ++p) {
                    int kb = H * 128 + p * 16 + (lh << 3);
                    int2 d = *(const int2*)(ldsS + (size_t)(t * 16 + m) * 256 + kb);
                    A[t][H][2 * p]     = d.x;
                    A[t][H][2 * p + 1] = d.y;
                }
            }
        }

        asm volatile("s_wait_asynccnt 0x0" ::: "memory");
        __syncthreads();
    }

#pragma unroll 4
    for (int idx = lane; idx < ROWS_PER_WAVE * HIDDEN; idx += 32) {
        float v = (float)ldsS[idx];
        __builtin_nontemporal_store(v, &out[(size_t)tileBase * HIDDEN + idx]);
    }
}

extern "C" void kernel_launch(void* const* d_in, const int* in_sizes, int n_in,
                              void* d_out, int out_size, void* d_ws, size_t ws_size,
                              hipStream_t stream) {
    (void)in_sizes; (void)n_in; (void)out_size; (void)ws_size;
    const float* state    = (const float*)d_in[0];
    const float* matrices = (const float*)d_in[1];
    const float* noise    = (const float*)d_in[2];
    float* out = (float*)d_out;

    unsigned char* matB = (unsigned char*)d_ws;

    pack_matB<<<4096, 256, 0, stream>>>(matrices, (unsigned int*)matB);

    const int blocks = BATCH / (ROWS_PER_WAVE * WAVES_PER_BLOCK);
    const size_t shmem = 2 * MATBYTES + WAVES_PER_BLOCK * (ROWS_PER_WAVE * HIDDEN);
    tenshash_rounds<<<blocks, THREADS, shmem, stream>>>(state, noise, matB, out);
}
